// TransformerBlock_4939212390582
// MI455X (gfx1250) — matrix operand order repaired and checked by reading, not compiled, not run on hardware
//
#include <hip/hip_runtime.h>

// ---------------------------------------------------------------------------
// Problem constants (from reference)
// ---------------------------------------------------------------------------
#define BB   16
#define DD   384
#define NN   4096      // H*W
#define DQ   768
#define DV   384
#define NHH  12
#define DFF  1536
#define HDQ  64        // Dq / NH
#define HDV  32        // Dv / NH

typedef __bf16 bf16;
typedef __attribute__((ext_vector_type(8)))  __bf16 v8bf;
typedef __attribute__((ext_vector_type(16))) __bf16 v16bf;
typedef __attribute__((ext_vector_type(8)))  float  v8f;

union V16U { v16bf v; v8bf h[2]; };

__device__ __forceinline__ v8f wmma_bf16(v16bf a, v16bf b, v8f c) {
  return __builtin_amdgcn_wmma_f32_16x16x32_bf16(false, a, false, b, (short)0, c,
                                                 false, false);
}
__device__ __forceinline__ v8f vz8() {
  v8f z; 
#pragma unroll
  for (int i = 0; i < 8; ++i) z[i] = 0.f;
  return z;
}

// ---------------------------------------------------------------------------
// Weight fp32 -> bf16 convert
// ---------------------------------------------------------------------------
__global__ void k_cvt(const float* __restrict__ s, bf16* __restrict__ d, int n) {
  int i = blockIdx.x * 256 + threadIdx.x;
  if (i < n) d[i] = (bf16)s[i];
}

// ---------------------------------------------------------------------------
// LayerNorm statistics: deterministic 2-stage reduction (no float atomics)
// grid (B, 48), block 256; per-batch element count = DD*NN = 1,572,864
// ---------------------------------------------------------------------------
__global__ void k_ln_part(const float* __restrict__ src, double* __restrict__ part) {
  const int per = (DD * NN) / 48;                 // 32768
  int b = blockIdx.x, ch = blockIdx.y;
  size_t base = (size_t)b * (DD * NN) + (size_t)ch * per;
  double s = 0.0, s2 = 0.0;
  for (int i = threadIdx.x; i < per; i += 256) {
    float v = src[base + i];
    s += v; s2 += (double)v * v;
  }
  __shared__ double ls[256], lq[256];
  ls[threadIdx.x] = s; lq[threadIdx.x] = s2;
  __syncthreads();
  for (int o = 128; o > 0; o >>= 1) {
    if (threadIdx.x < o) { ls[threadIdx.x] += ls[threadIdx.x + o];
                           lq[threadIdx.x] += lq[threadIdx.x + o]; }
    __syncthreads();
  }
  if (threadIdx.x == 0) {
    part[(b * 48 + ch) * 2 + 0] = ls[0];
    part[(b * 48 + ch) * 2 + 1] = lq[0];
  }
}

__global__ void k_ln_final(const double* __restrict__ part, float* __restrict__ stats) {
  int b = threadIdx.x;
  if (b >= BB) return;
  double s = 0.0, s2 = 0.0;
  for (int c = 0; c < 48; ++c) { s += part[(b * 48 + c) * 2]; s2 += part[(b * 48 + c) * 2 + 1]; }
  double n = (double)DD * NN;
  double mu = s / n;
  double var = s2 / n - mu * mu;
  stats[b * 2 + 0] = (float)mu;
  stats[b * 2 + 1] = (float)(1.0 / sqrt(var + 1e-5));
}

// ---------------------------------------------------------------------------
// LN1 apply: x (B,C,N) f32 -> a_bf (B,N,C) bf16, LDS-tiled transpose
// grid (NN/64, DD/64, B), block 256
// ---------------------------------------------------------------------------
__global__ __launch_bounds__(256) void k_ln1_apply(
    const float* __restrict__ x, const float* __restrict__ w,
    const float* __restrict__ bb, const float* __restrict__ stats,
    bf16* __restrict__ a) {
  __shared__ float lx[64][65], lw[64][65], lb[64][65];
  int b = blockIdx.z, c0 = blockIdx.y * 64, n0 = blockIdx.x * 64;
  int tx = threadIdx.x & 63, ty = threadIdx.x >> 6;
  float mu = stats[b * 2], rs = stats[b * 2 + 1];
#pragma unroll
  for (int r = 0; r < 16; ++r) {
    int cl = r * 4 + ty;
    size_t ci = (size_t)(c0 + cl) * NN + n0 + tx;
    lx[cl][tx] = x[(size_t)b * (DD * NN) + ci];
    lw[cl][tx] = w[ci];
    lb[cl][tx] = bb[ci];
  }
  __syncthreads();
#pragma unroll
  for (int r = 0; r < 16; ++r) {
    int nl = r * 4 + ty, cl = tx;
    float v = (lx[cl][nl] - mu) * rs * lw[cl][nl] + lb[cl][nl];
    a[((size_t)(b * NN + n0 + nl)) * DD + c0 + cl] = (bf16)v;
  }
}

// ---------------------------------------------------------------------------
// LN2 apply: x2 (B,N,C) f32 -> h f32 (B,C,N) into d_out  +  h_bf (B,N,C)
// ---------------------------------------------------------------------------
__global__ __launch_bounds__(256) void k_ln2_apply(
    const float* __restrict__ x2, const float* __restrict__ w,
    const float* __restrict__ bb, const float* __restrict__ stats,
    float* __restrict__ hout_cf, bf16* __restrict__ hbf) {
  __shared__ float lw[64][65], lb[64][65], lh[64][65];
  int b = blockIdx.z, c0 = blockIdx.y * 64, n0 = blockIdx.x * 64;
  int tx = threadIdx.x & 63, ty = threadIdx.x >> 6;
  float mu = stats[b * 2], rs = stats[b * 2 + 1];
#pragma unroll
  for (int r = 0; r < 16; ++r) {
    int cl = r * 4 + ty;
    size_t ci = (size_t)(c0 + cl) * NN + n0 + tx;
    lw[cl][tx] = w[ci];
    lb[cl][tx] = bb[ci];
  }
  __syncthreads();
#pragma unroll
  for (int r = 0; r < 16; ++r) {
    int nl = r * 4 + ty, cl = tx;
    size_t xi = ((size_t)(b * NN + n0 + nl)) * DD + c0 + cl;
    float v = (x2[xi] - mu) * rs * lw[cl][nl] + lb[cl][nl];
    hbf[xi] = (bf16)v;
    lh[cl][nl] = v;
  }
  __syncthreads();
#pragma unroll
  for (int r = 0; r < 16; ++r) {
    int cl = r * 4 + ty, nl = tx;
    hout_cf[(size_t)b * (DD * NN) + (size_t)(c0 + cl) * NN + n0 + nl] = lh[cl][nl];
  }
}

// ---------------------------------------------------------------------------
// Main bf16 WMMA GEMM: out(Cout,N) = W(Cout,Cin) x act(N,Cin)^T  per batch.
// Block 128x256 tile, 8 waves, 64x64 per wave (16 wmma tiles). Fragments are
// loaded directly from global with b128 loads (K-contiguous layouts).
// grid (NN/256, Cout/128, B), block 256.
// ---------------------------------------------------------------------------
__global__ __launch_bounds__(256) void k_gemm(
    const bf16* __restrict__ W, const bf16* __restrict__ act,
    const float* __restrict__ bias, const float* __restrict__ res_cf,
    float* __restrict__ outf, bf16* __restrict__ outb,
    int Cin, int Cout, int out_cf, int do_gelu) {
  int b = blockIdx.z;
  int wave = threadIdx.x >> 5, lane = threadIdx.x & 31;
  int l = lane & 15, hi = lane >> 4;
  int wm = wave & 1, wn = wave >> 1;
  int mBase = blockIdx.y * 128 + wm * 64;
  int nBase = blockIdx.x * 256 + wn * 64;
  const bf16* actb = act + (size_t)b * NN * Cin;

  v8f acc[4][4];
#pragma unroll
  for (int i = 0; i < 4; ++i)
#pragma unroll
    for (int j = 0; j < 4; ++j) acc[i][j] = vz8();

  for (int k0 = 0; k0 < Cin; k0 += 32) {
    v16bf af[4], bfr[4];
#pragma unroll
    for (int mt = 0; mt < 4; ++mt) {
      const bf16* p = W + (size_t)(mBase + mt * 16 + l) * Cin + k0 + hi * 8;
      V16U u; u.h[0] = *(const v8bf*)p; u.h[1] = *(const v8bf*)(p + 16);
      af[mt] = u.v;
    }
#pragma unroll
    for (int nt = 0; nt < 4; ++nt) {
      const bf16* p = actb + (size_t)(nBase + nt * 16 + l) * Cin + k0 + hi * 8;
      V16U u; u.h[0] = *(const v8bf*)p; u.h[1] = *(const v8bf*)(p + 16);
      bfr[nt] = u.v;
    }
#pragma unroll
    for (int mt = 0; mt < 4; ++mt)
#pragma unroll
      for (int nt = 0; nt < 4; ++nt)
        acc[mt][nt] = wmma_bf16(af[mt], bfr[nt], acc[mt][nt]);
  }

#pragma unroll
  for (int mt = 0; mt < 4; ++mt)
#pragma unroll
    for (int nt = 0; nt < 4; ++nt) {
      int n = nBase + nt * 16 + l;
#pragma unroll
      for (int r = 0; r < 8; ++r) {
        int m = mBase + mt * 16 + hi * 8 + r;
        float v = acc[mt][nt][r] + bias[m];
        if (do_gelu) v = 0.5f * v * (1.f + erff(v * 0.70710678118654752f));
        if (res_cf) v += res_cf[(size_t)(b * Cout + m) * NN + n];
        if (out_cf) {
          if (outf) outf[(size_t)(b * Cout + m) * NN + n] = v;
        } else {
          size_t o = (size_t)(b * NN + n) * Cout + m;
          if (outf) outf[o] = v;
          if (outb) outb[o] = (bf16)v;
        }
      }
    }
}

// ---------------------------------------------------------------------------
// Q channel softmax: one wave per pixel over 768 channels (contiguous).
// grid (B*NN/8), block 256
// ---------------------------------------------------------------------------
__global__ __launch_bounds__(256) void k_qsm(const float* __restrict__ Qf,
                                             bf16* __restrict__ Qs) {
  int pix = blockIdx.x * 8 + (threadIdx.x >> 5);
  int lane = threadIdx.x & 31;
  const float* p = Qf + (size_t)pix * DQ;
  float v[24];
  float m = -3.4e38f;
#pragma unroll
  for (int i = 0; i < 24; ++i) { v[i] = p[lane + i * 32]; m = fmaxf(m, v[i]); }
  for (int o = 16; o > 0; o >>= 1) m = fmaxf(m, __shfl_xor(m, o));
  float s = 0.f;
#pragma unroll
  for (int i = 0; i < 24; ++i) { v[i] = expf(v[i] - m); s += v[i]; }
  for (int o = 16; o > 0; o >>= 1) s += __shfl_xor(s, o);
  float inv = 1.f / s;
  bf16* q = Qs + (size_t)pix * DQ;
#pragma unroll
  for (int i = 0; i < 24; ++i) q[lane + i * 32] = (bf16)(v[i] * inv);
}

// ---------------------------------------------------------------------------
// K spatial softmax (over N=4096 per (b,c)): split online pass + merge + write
// ---------------------------------------------------------------------------
__global__ void k_ksm_part(const float* __restrict__ Kf, float* __restrict__ pmax,
                           float* __restrict__ psum) {
  int b = blockIdx.x, ch = blockIdx.y, c = threadIdx.x;   // blockDim = 768
  float m = -3.4e38f, s = 0.f;
  for (int j = 0; j < 512; ++j) {
    int n = ch * 512 + j;
    float v = Kf[((size_t)(b * NN + n)) * DQ + c];
    float m2 = fmaxf(m, v);
    s = s * expf(m - m2) + expf(v - m2);
    m = m2;
  }
  pmax[(b * 8 + ch) * DQ + c] = m;
  psum[(b * 8 + ch) * DQ + c] = s;
}

__global__ void k_ksm_merge(const float* __restrict__ pmax,
                            const float* __restrict__ psum, float* __restrict__ gms) {
  int i = blockIdx.x * 256 + threadIdx.x;
  if (i >= BB * DQ) return;
  int b = i / DQ, c = i % DQ;
  float M = -3.4e38f;
  for (int ch = 0; ch < 8; ++ch) M = fmaxf(M, pmax[(b * 8 + ch) * DQ + c]);
  float S = 0.f;
  for (int ch = 0; ch < 8; ++ch) S += psum[(b * 8 + ch) * DQ + c] * expf(pmax[(b * 8 + ch) * DQ + c] - M);
  gms[i * 2 + 0] = M;
  gms[i * 2 + 1] = 1.f / S;
}

__global__ void k_ksm_write(const float* __restrict__ Kf, const float* __restrict__ gms,
                            bf16* __restrict__ Ks) {
  int b = blockIdx.x, c = threadIdx.x;                    // blockDim = 768
  int n0 = blockIdx.y * 128;
  float M = gms[(b * DQ + c) * 2], inv = gms[(b * DQ + c) * 2 + 1];
  for (int j = 0; j < 128; ++j) {
    size_t idx = ((size_t)(b * NN + n0 + j)) * DQ + c;
    Ks[idx] = (bf16)(expf(Kf[idx] - M) * inv);
  }
}

// ---------------------------------------------------------------------------
// KV einsum: KV[b,h,q,v] = (1/NN) * sum_n Ks[b,n,h*64+q] * V[b,n,h*32+v]
// WMMA with K = n; LDS transpose staging. grid (B*NH), block 64 (2 waves).
// ---------------------------------------------------------------------------
__global__ __launch_bounds__(64) void k_kv(const bf16* __restrict__ Ks,
                                           const bf16* __restrict__ Vb,
                                           float* __restrict__ KV) {
  int b = blockIdx.x / NHH, h = blockIdx.x % NHH;
  __shared__ __align__(16) bf16 sK[64][40];   // [q][n_local]
  __shared__ __align__(16) bf16 sV[32][40];   // [v][n_local]
  int tid = threadIdx.x, wave = tid >> 5, lane = tid & 31;
  int l = lane & 15, hi = lane >> 4;
  v8f acc[2][2];
#pragma unroll
  for (int i = 0; i < 2; ++i)
#pragma unroll
    for (int j = 0; j < 2; ++j) acc[i][j] = vz8();

  for (int n0 = 0; n0 < NN; n0 += 32) {
    if (tid < 32) {
      const bf16* p = Ks + ((size_t)(b * NN + n0 + tid)) * DQ + h * HDQ;
#pragma unroll
      for (int q = 0; q < 64; ++q) sK[q][tid] = p[q];
    } else {
      const bf16* p = Vb + ((size_t)(b * NN + n0 + tid - 32)) * DV + h * HDV;
#pragma unroll
      for (int vv = 0; vv < 32; ++vv) sV[vv][tid - 32] = p[vv];
    }
    __syncthreads();
    v16bf af[2], bfr[2];
#pragma unroll
    for (int mt = 0; mt < 2; ++mt) {
      const bf16* p = &sK[wave * 32 + mt * 16 + l][0];
      V16U u; u.h[0] = *(const v8bf*)(p + hi * 8); u.h[1] = *(const v8bf*)(p + 16 + hi * 8);
      af[mt] = u.v;
    }
#pragma unroll
    for (int nt = 0; nt < 2; ++nt) {
      const bf16* p = &sV[nt * 16 + l][0];
      V16U u; u.h[0] = *(const v8bf*)(p + hi * 8); u.h[1] = *(const v8bf*)(p + 16 + hi * 8);
      bfr[nt] = u.v;
    }
#pragma unroll
    for (int mt = 0; mt < 2; ++mt)
#pragma unroll
      for (int nt = 0; nt < 2; ++nt)
        acc[mt][nt] = wmma_bf16(af[mt], bfr[nt], acc[mt][nt]);
    __syncthreads();
  }
  const float sc = 1.f / (float)NN;
#pragma unroll
  for (int mt = 0; mt < 2; ++mt)
#pragma unroll
    for (int nt = 0; nt < 2; ++nt)
#pragma unroll
      for (int r = 0; r < 8; ++r) {
        int q = wave * 32 + mt * 16 + hi * 8 + r;
        int vv = nt * 16 + l;
        KV[(((size_t)b * NHH + h) * HDQ + q) * HDV + vv] = acc[mt][nt][r] * sc;
      }
}

// ---------------------------------------------------------------------------
// QKV: out[b,n,h*32+v] = sum_q KV[b,h,q,v] * Qs[b,n,h*64+q]   (K = 64)
// A = KV^T staged in LDS (bf16); B fragments straight from Qs (K-contig).
// grid (B, NH, NN/512), block 256 (8 waves x 64 pixels each).
// ---------------------------------------------------------------------------
__global__ __launch_bounds__(256) void k_qkv(const float* __restrict__ KV,
                                             const bf16* __restrict__ Qs,
                                             bf16* __restrict__ outb) {
  int b = blockIdx.x, h = blockIdx.y;
  __shared__ __align__(16) bf16 sA[32][64];   // [v][q]
  int tid = threadIdx.x;
  for (int i = tid; i < 32 * 64; i += 256) {
    int vv = i >> 6, q = i & 63;
    sA[vv][q] = (bf16)KV[(((size_t)b * NHH + h) * HDQ + q) * HDV + vv];
  }
  __syncthreads();
  int wave = tid >> 5, lane = tid & 31, l = lane & 15, hi = lane >> 4;
  int nBase = blockIdx.z * 512 + wave * 64;
  v8f acc[2][4];
#pragma unroll
  for (int i = 0; i < 2; ++i)
#pragma unroll
    for (int j = 0; j < 4; ++j) acc[i][j] = vz8();

#pragma unroll
  for (int k0 = 0; k0 < 64; k0 += 32) {
    v16bf af[2], bfr[4];
#pragma unroll
    for (int mt = 0; mt < 2; ++mt) {
      const bf16* p = &sA[mt * 16 + l][k0];
      V16U u; u.h[0] = *(const v8bf*)(p + hi * 8); u.h[1] = *(const v8bf*)(p + 16 + hi * 8);
      af[mt] = u.v;
    }
#pragma unroll
    for (int nt = 0; nt < 4; ++nt) {
      const bf16* p = Qs + ((size_t)(b * NN + nBase + nt * 16 + l)) * DQ + h * HDQ + k0 + hi * 8;
      V16U u; u.h[0] = *(const v8bf*)p; u.h[1] = *(const v8bf*)(p + 16);
      bfr[nt] = u.v;
    }
#pragma unroll
    for (int mt = 0; mt < 2; ++mt)
#pragma unroll
      for (int nt = 0; nt < 4; ++nt)
        acc[mt][nt] = wmma_bf16(af[mt], bfr[nt], acc[mt][nt]);
  }
#pragma unroll
  for (int mt = 0; mt < 2; ++mt)
#pragma unroll
    for (int nt = 0; nt < 4; ++nt)
#pragma unroll
      for (int r = 0; r < 8; ++r) {
        int vv = mt * 16 + hi * 8 + r;
        int n = nBase + nt * 16 + l;
        outb[((size_t)(b * NN + n)) * DV + h * HDV + vv] = (bf16)acc[mt][nt][r];
      }
}

// ---------------------------------------------------------------------------
// Workspace arena (byte offsets, sequential stages allow aliasing)
// ---------------------------------------------------------------------------
static constexpr size_t O_WQ   = 0;
static constexpr size_t O_WK   = O_WQ  + (size_t)DQ  * DD  * 2;
static constexpr size_t O_WV   = O_WK  + (size_t)DQ  * DD  * 2;
static constexpr size_t O_WHU  = O_WV  + (size_t)DV  * DD  * 2;
static constexpr size_t O_WF1  = O_WHU + (size_t)DD  * DV  * 2;
static constexpr size_t O_WF2  = O_WF1 + (size_t)DFF * DD  * 2;
static constexpr size_t O_ST   = O_WF2 + (size_t)DD  * DFF * 2;     // stats1 | stats2
static constexpr size_t O_PART = O_ST   + 256;                      // 16*48*2 doubles
static constexpr size_t O_PMAX = O_PART + (size_t)BB * 48 * 2 * 8;
static constexpr size_t O_PSUM = O_PMAX + (size_t)BB * 8 * DQ * 4;
static constexpr size_t O_GMS  = O_PSUM + (size_t)BB * 8 * DQ * 4;
static constexpr size_t O_ABF  = (O_GMS + (size_t)BB * DQ * 2 * 4 + 255) & ~(size_t)255;
static constexpr size_t SZ_ABF = (size_t)BB * NN * DD * 2;          // 50.3 MB
static constexpr size_t SZ_QF  = (size_t)BB * NN * DQ * 4;          // 201 MB
static constexpr size_t SZ_QS  = (size_t)BB * NN * DQ * 2;          // 100 MB
static constexpr size_t O_QF   = O_ABF + SZ_ABF;
static constexpr size_t O_KF   = O_QF  + SZ_QF;
static constexpr size_t O_VB   = O_KF  + SZ_QF;
static constexpr size_t O_QS   = O_VB  + SZ_ABF;
static constexpr size_t O_KS   = O_QS  + SZ_QS;
static constexpr size_t O_KV   = O_KS  + SZ_QS;
static constexpr size_t O_END  = O_KV  + (size_t)BB * NHH * HDQ * HDV * 4;
// aliases (dead buffers reused by later stages)
static constexpr size_t O_QKVB = O_ABF;   // a_bf dead after V GEMM
static constexpr size_t O_X2   = O_KF;    // Kf dead after K-softmax write
static constexpr size_t O_GBF  = O_QF;    // Qf dead after Q-softmax
static constexpr size_t O_HBF  = O_KS;    // Ks dead after KV einsum

extern "C" void kernel_launch(void* const* d_in, const int* in_sizes, int n_in,
                              void* d_out, int out_size, void* d_ws, size_t ws_size,
                              hipStream_t stream) {
  (void)in_sizes; (void)n_in; (void)out_size;
  if (ws_size < O_END) return;

  const float* x    = (const float*)d_in[0];
  const float* ln1w = (const float*)d_in[1];
  const float* ln1b = (const float*)d_in[2];
  const float* qw   = (const float*)d_in[3];
  const float* qb   = (const float*)d_in[4];
  const float* kw   = (const float*)d_in[5];
  const float* kb   = (const float*)d_in[6];
  const float* vw   = (const float*)d_in[7];
  const float* vb   = (const float*)d_in[8];
  const float* huw  = (const float*)d_in[9];
  const float* hub  = (const float*)d_in[10];
  const float* ln2w = (const float*)d_in[11];
  const float* ln2b = (const float*)d_in[12];
  const float* f1w  = (const float*)d_in[13];
  const float* f1b  = (const float*)d_in[14];
  const float* f2w  = (const float*)d_in[15];
  const float* f2b  = (const float*)d_in[16];
  float* out = (float*)d_out;
  char*  ws  = (char*)d_ws;

  bf16*   WQ   = (bf16*)(ws + O_WQ);
  bf16*   WK   = (bf16*)(ws + O_WK);
  bf16*   WV   = (bf16*)(ws + O_WV);
  bf16*   WHU  = (bf16*)(ws + O_WHU);
  bf16*   WF1  = (bf16*)(ws + O_WF1);
  bf16*   WF2  = (bf16*)(ws + O_WF2);
  float*  st1  = (float*)(ws + O_ST);
  float*  st2  = (float*)(ws + O_ST + 128);
  double* part = (double*)(ws + O_PART);
  float*  pmax = (float*)(ws + O_PMAX);
  float*  psum = (float*)(ws + O_PSUM);
  float*  gms  = (float*)(ws + O_GMS);
  bf16*   ABF  = (bf16*)(ws + O_ABF);
  float*  QF   = (float*)(ws + O_QF);
  float*  KF   = (float*)(ws + O_KF);
  bf16*   VB   = (bf16*)(ws + O_VB);
  bf16*   QS   = (bf16*)(ws + O_QS);
  bf16*   KS   = (bf16*)(ws + O_KS);
  float*  KVp  = (float*)(ws + O_KV);
  bf16*   QKVB = (bf16*)(ws + O_QKVB);
  float*  X2   = (float*)(ws + O_X2);
  bf16*   GBF  = (bf16*)(ws + O_GBF);
  bf16*   HBF  = (bf16*)(ws + O_HBF);

  // 1. convert weights to bf16
  k_cvt<<<(DQ * DD + 255) / 256, 256, 0, stream>>>(qw, WQ, DQ * DD);
  k_cvt<<<(DQ * DD + 255) / 256, 256, 0, stream>>>(kw, WK, DQ * DD);
  k_cvt<<<(DV * DD + 255) / 256, 256, 0, stream>>>(vw, WV, DV * DD);
  k_cvt<<<(DD * DV + 255) / 256, 256, 0, stream>>>(huw, WHU, DD * DV);
  k_cvt<<<(DFF * DD + 255) / 256, 256, 0, stream>>>(f1w, WF1, DFF * DD);
  k_cvt<<<(DD * DFF + 255) / 256, 256, 0, stream>>>(f2w, WF2, DD * DFF);

  // 2. LN1
  k_ln_part<<<dim3(BB, 48), 256, 0, stream>>>(x, part);
  k_ln_final<<<1, 32, 0, stream>>>(part, st1);
  k_ln1_apply<<<dim3(NN / 64, DD / 64, BB), 256, 0, stream>>>(x, ln1w, ln1b, st1, ABF);

  // 3. Q / K / V projections (bf16 WMMA GEMMs)
  k_gemm<<<dim3(NN / 256, DQ / 128, BB), 256, 0, stream>>>(WQ, ABF, qb, nullptr, QF, nullptr, DD, DQ, 0, 0);
  k_gemm<<<dim3(NN / 256, DQ / 128, BB), 256, 0, stream>>>(WK, ABF, kb, nullptr, KF, nullptr, DD, DQ, 0, 0);
  k_gemm<<<dim3(NN / 256, DV / 128, BB), 256, 0, stream>>>(WV, ABF, vb, nullptr, nullptr, VB, DD, DV, 0, 0);

  // 4. softmaxes
  k_qsm<<<(BB * NN) / 8, 256, 0, stream>>>(QF, QS);
  k_ksm_part<<<dim3(BB, 8), DQ, 0, stream>>>(KF, pmax, psum);
  k_ksm_merge<<<(BB * DQ + 255) / 256, 256, 0, stream>>>(pmax, psum, gms);
  k_ksm_write<<<dim3(BB, 32), DQ, 0, stream>>>(KF, gms, KS);

  // 5. per-head einsums (WMMA)
  k_kv<<<BB * NHH, 64, 0, stream>>>(KS, VB, KVp);
  k_qkv<<<dim3(BB, NHH, NN / 512), 256, 0, stream>>>(KVp, QS, QKVB);

  // 6. head unification + residual -> x2 (channels-last f32)
  k_gemm<<<dim3(NN / 256, DD / 128, BB), 256, 0, stream>>>(WHU, QKVB, hub, x, X2, nullptr, DV, DD, 0, 0);

  // 7. LN2 -> h (d_out, channels-first) + h_bf (channels-last)
  k_ln_part<<<dim3(BB, 48), 256, 0, stream>>>(X2, part);
  k_ln_final<<<1, 32, 0, stream>>>(part, st2);
  k_ln2_apply<<<dim3(NN / 64, DD / 64, BB), 256, 0, stream>>>(X2, ln2w, ln2b, st2, out, HBF);

  // 8. FFN: ff1 + exact GELU fused, ff2 + final residual fused into d_out
  k_gemm<<<dim3(NN / 256, DFF / 128, BB), 256, 0, stream>>>(WF1, HBF, f1b, nullptr, nullptr, GBF, DD, DFF, 0, 1);
  k_gemm<<<dim3(NN / 256, DD / 128, BB), 256, 0, stream>>>(WF2, GBF, f2b, out, out, nullptr, DFF, DD, 1, 0);
}
